// SelfAttention_67216238182590
// MI455X (gfx1250) — matrix operand order repaired and checked by reading, not compiled, not run on hardware
//
#include <hip/hip_runtime.h>
#include <hip/hip_bf16.h>

// ---------------------------------------------------------------------------
// Self-attention (single "head" of width 768) for MI455X / gfx1250, wave32.
//   K0: Wq/Wk/Wv fp32 -> bf16 (once; weights then L2-resident for all blocks)
//   K1: Q = (x@Wq^T)/sqrt(768), K = x@Wk^T, Vt = (x@Wv^T)^T  (bf16, in ws)
//   K2: scores = Q K^T ; softmax ; H = P V   (all via v_wmma_f32_16x16x32_bf16)
// Workspace: Qb|Kb|Vt (24 MB each) + Wb (3.5 MB) = 79,036,416 B.
// ---------------------------------------------------------------------------

#define EMBED 768
#define SEQ   2048
#define BATCH 8
#define NROWS (BATCH * SEQ) // 16384
#define WMAT  (EMBED * EMBED)

typedef __bf16        bf16x16 __attribute__((ext_vector_type(16)));
typedef __bf16        bf16x8  __attribute__((ext_vector_type(8)));
typedef float         fx8     __attribute__((ext_vector_type(8)));
typedef float         fx4     __attribute__((ext_vector_type(4)));
typedef unsigned int  ux4     __attribute__((ext_vector_type(4)));

// Native converts: lowers to v_cvt_pk_bf16_f32 on gfx1250.
__device__ __forceinline__ __bf16 f2bf(float f) { return (__bf16)f; }
__device__ __forceinline__ float bf2f(__bf16 b) {
    unsigned short h = __builtin_bit_cast(unsigned short, b);
    return __uint_as_float(((unsigned)h) << 16);
}
__device__ __forceinline__ unsigned short bfbits(__bf16 b) {
    return __builtin_bit_cast(unsigned short, b);
}
__device__ __forceinline__ bf16x16 cat16(bf16x8 lo, bf16x8 hi) {
    bf16x16 r;
#pragma unroll
    for (int i = 0; i < 8; ++i) { r[i] = lo[i]; r[i + 8] = hi[i]; }
    return r;
}

// A-fragment (16x32 bf16, MxK): lane = half*16 + M.
//   e=0..7  -> K = kbase + half*8 + e ; e=8..15 -> K = kbase + 16 + half*8 + (e-8)
// Two 16B-aligned contiguous runs -> ds_load_b128 / global_load_b128.
__device__ __forceinline__ bf16x16 loadA(const __bf16* base, int stride,
                                         int rowM, int kbase, int half) {
    const bf16x8* p0 = (const bf16x8*)(base + (size_t)rowM * stride + kbase + half * 8);
    const bf16x8* p1 = (const bf16x8*)(base + (size_t)rowM * stride + kbase + 16 + half * 8);
    return cat16(*p0, *p1);
}
// B-fragment (32x16 bf16, KxN): lane = half*16 + N, same K order as the A-fragment.
__device__ __forceinline__ bf16x16 loadB(const __bf16* rowN_base, int kbase, int half) {
    const bf16x8* p0 = (const bf16x8*)(rowN_base + kbase + half * 8);
    const bf16x8* p1 = (const bf16x8*)(rowN_base + kbase + 16 + half * 8);
    return cat16(*p0, *p1);
}

// ---------------------------------------------------------------------------
// Kernel 0: one-shot fp32 -> bf16 weight conversion (3 x 768 x 768).
// 864 blocks x 256 threads, 8 elements per thread (b128 in, b128 out).
// ---------------------------------------------------------------------------
__global__ __launch_bounds__(256) void wconv_kernel(
    const float* __restrict__ Wq,
    const float* __restrict__ Wk,
    const float* __restrict__ Wv,
    unsigned short* __restrict__ Wb_)
{
    __bf16* Wb = (__bf16*)Wb_;
    const size_t g   = ((size_t)blockIdx.x * 256 + threadIdx.x) * 8;
    const int    mat = (int)(g / WMAT);
    const size_t off = g - (size_t)mat * WMAT;
    const float* src = (mat == 0) ? Wq : ((mat == 1) ? Wk : Wv);
    fx4 a = *(const fx4*)(src + off);
    fx4 b = *(const fx4*)(src + off + 4);
    bf16x8 o;
#pragma unroll
    for (int j = 0; j < 4; ++j) { o[j] = f2bf(a[j]); o[j + 4] = f2bf(b[j]); }
    *(bf16x8*)(Wb + g) = o;
}

// ---------------------------------------------------------------------------
// Kernel 1: QKV projection. 1024 blocks x 128 threads (4 waves).
// Block owns 16 token rows; each wave owns 12 of the 48 feature tiles and
// computes Q, K, V for that tile together (one A-frag feeds three WMMAs).
// ---------------------------------------------------------------------------
__global__ __launch_bounds__(128) void qkv_proj_kernel(
    const float* __restrict__ x,
    const unsigned short* __restrict__ Wb_,
    unsigned short* __restrict__ Qb_,
    unsigned short* __restrict__ Kb_,
    unsigned short* __restrict__ Vt_)
{
    __shared__ __align__(16) __bf16 Xl[16 * EMBED];   // 24 KB bf16 x-tile

    const int tid   = threadIdx.x;
    const int wave  = tid >> 5;
    const int lane  = tid & 31;
    const int laneN = lane & 15;
    const int half  = lane >> 4;
    const int r0    = blockIdx.x * 16;                // first global token row

    // Stage the contiguous 16x768 fp32 tile as bf16 into LDS.
    const float* xs = x + (size_t)r0 * EMBED;
    for (int i = tid; i < (16 * EMBED) / 4; i += 128) {
        fx4 t = ((const fx4*)xs)[i];
#pragma unroll
        for (int j = 0; j < 4; ++j) Xl[i * 4 + j] = f2bf(t[j]);
    }
    __syncthreads();

    const __bf16* Wb = (const __bf16*)Wb_;
    __bf16* Qb = (__bf16*)Qb_;
    __bf16* Kb = (__bf16*)Kb_;
    __bf16* Vt = (__bf16*)Vt_;
    const float qscale = 0.036084391824351615f;       // 1/sqrt(768)
    const int   bb = r0 >> 11, s0 = r0 & 2047;

    for (int c = wave; c < 48; c += 4) {
        const int nb = c * 16;
        const __bf16* wq = Wb + 0 * (size_t)WMAT + (size_t)(nb + laneN) * EMBED;
        const __bf16* wk = Wb + 1 * (size_t)WMAT + (size_t)(nb + laneN) * EMBED;
        const __bf16* wv = Wb + 2 * (size_t)WMAT + (size_t)(nb + laneN) * EMBED;

        fx8 aq = {0.f,0.f,0.f,0.f,0.f,0.f,0.f,0.f};
        fx8 ak = {0.f,0.f,0.f,0.f,0.f,0.f,0.f,0.f};
        fx8 av = {0.f,0.f,0.f,0.f,0.f,0.f,0.f,0.f};

#pragma unroll 4
        for (int ks = 0; ks < 24; ++ks) {             // K = 768 in steps of 32
            const int kb = ks * 32;
            bf16x16 A  = loadA(Xl, EMBED, laneN, kb, half);
            bf16x16 Bq = loadB(wq, kb, half);
            bf16x16 Bk = loadB(wk, kb, half);
            bf16x16 Bv = loadB(wv, kb, half);
            aq = __builtin_amdgcn_wmma_f32_16x16x32_bf16(false, A, false, Bq,
                                                         (short)0, aq, false, false);
            ak = __builtin_amdgcn_wmma_f32_16x16x32_bf16(false, A, false, Bk,
                                                         (short)0, ak, false, false);
            av = __builtin_amdgcn_wmma_f32_16x16x32_bf16(false, A, false, Bv,
                                                         (short)0, av, false, false);
        }

        // Q, K: row-major bf16 (scatter b16 along rows).
#pragma unroll
        for (int v = 0; v < 8; ++v) {
            const int M = v + 8 * half;
            Qb[(size_t)(r0 + M) * EMBED + nb + laneN] = f2bf(aq[v] * qscale);
            Kb[(size_t)(r0 + M) * EMBED + nb + laneN] = f2bf(ak[v]);
        }
        // V transposed: consecutive v -> consecutive s, pack pairs into b32.
        unsigned* vdst = (unsigned*)(Vt + (size_t)bb * EMBED * SEQ
                                        + (size_t)(nb + laneN) * SEQ + s0 + 8 * half);
#pragma unroll
        for (int v = 0; v < 8; v += 2) {
            unsigned lo = bfbits(f2bf(av[v]));
            unsigned hi = bfbits(f2bf(av[v + 1]));
            vdst[v >> 1] = lo | (hi << 16);
        }
    }
}

// ---------------------------------------------------------------------------
// Kernel 2: attention for one 16-row q-tile per block. 1024 blocks x 128 thr.
// Dynamic LDS: P/scores bf16[16][2048] (64K) + Q bf16[16][768] (24K) + 640B red.
// ---------------------------------------------------------------------------
#define ATTN_LDS_BYTES (16 * SEQ * 2 + 16 * EMBED * 2 + 160 * 4)  // 90752

__global__ __launch_bounds__(128) void attn_kernel(
    const unsigned short* __restrict__ Qb_,
    const unsigned short* __restrict__ Kb_,
    const unsigned short* __restrict__ Vt_,
    float* __restrict__ out)
{
    extern __shared__ __align__(16) char smem[];
    __bf16* Pl     = (__bf16*)smem;                               // 16*2048 bf16
    __bf16* Ql     = (__bf16*)(smem + 16 * SEQ * 2);              // 16*768 bf16
    float*  red    = (float*)(smem + 16 * SEQ * 2 + 16 * EMBED * 2); // [16][8]
    float*  rowmax = red + 128;                                   // [16]
    float*  rowinv = rowmax + 16;                                 // [16]

    const int tid   = threadIdx.x;
    const int wave  = tid >> 5;
    const int lane  = tid & 31;
    const int laneN = lane & 15;
    const int half  = lane >> 4;
    const int r0    = blockIdx.x * 16;
    const int b     = r0 >> 11;

    const __bf16* Qb = (const __bf16*)Qb_;
    const __bf16* Kb = (const __bf16*)Kb_;
    const __bf16* Vt = (const __bf16*)Vt_;

    // Stage Q tile (contiguous 16*768 bf16 = 24 KB) into LDS.
    {
        const ux4* src = (const ux4*)(Qb + (size_t)r0 * EMBED);
        ux4*       dst = (ux4*)Ql;
        for (int i = tid; i < (16 * EMBED * 2) / 16; i += 128) dst[i] = src[i];
    }
    __syncthreads();

    // ---- Phase 1: scores = Q K^T, two key-tiles per A-frag load.
    const __bf16* Kbase = Kb + (size_t)b * SEQ * EMBED;
    for (int jt = wave * 2; jt < SEQ / 16; jt += 8) {             // jt, jt+1
        fx8 acc0 = {0.f,0.f,0.f,0.f,0.f,0.f,0.f,0.f};
        fx8 acc1 = {0.f,0.f,0.f,0.f,0.f,0.f,0.f,0.f};
        const __bf16* krow0 = Kbase + (size_t)(jt * 16 + laneN) * EMBED;
        const __bf16* krow1 = krow0 + (size_t)16 * EMBED;
        __builtin_prefetch(krow0, 0, 1);
#pragma unroll 4
        for (int ks = 0; ks < 24; ++ks) {
            bf16x16 A  = loadA(Ql, EMBED, laneN, ks * 32, half);
            bf16x16 B0 = loadB(krow0, ks * 32, half);
            bf16x16 B1 = loadB(krow1, ks * 32, half);
            acc0 = __builtin_amdgcn_wmma_f32_16x16x32_bf16(false, A, false, B0,
                                                           (short)0, acc0, false, false);
            acc1 = __builtin_amdgcn_wmma_f32_16x16x32_bf16(false, A, false, B1,
                                                           (short)0, acc1, false, false);
        }
#pragma unroll
        for (int v = 0; v < 8; ++v) {
            const size_t row = (size_t)(v + 8 * half) * SEQ;
            Pl[row + jt * 16 + laneN]      = f2bf(acc0[v]);
            Pl[row + jt * 16 + 16 + laneN] = f2bf(acc1[v]);
        }
    }
    __syncthreads();

    // ---- Softmax statistics: row max, then in-place P = exp(s - max), row sums.
    {
        const int row = tid >> 3, seg = tid & 7;
        const bf16x8* p = (const bf16x8*)(Pl + row * SEQ + seg * (SEQ / 8));
        float m = -3.0e38f;
        for (int c = 0; c < SEQ / 8 / 8; ++c) {
            bf16x8 t = p[c];
#pragma unroll
            for (int j = 0; j < 8; ++j) m = fmaxf(m, bf2f(t[j]));
        }
        red[row * 8 + seg] = m;
    }
    __syncthreads();
    if (tid < 16) {
        float m = red[tid * 8];
#pragma unroll
        for (int j = 1; j < 8; ++j) m = fmaxf(m, red[tid * 8 + j]);
        rowmax[tid] = m;
    }
    __syncthreads();
    {
        const int row = tid >> 3, seg = tid & 7;
        bf16x8* p = (bf16x8*)(Pl + row * SEQ + seg * (SEQ / 8));
        const float m = rowmax[row];
        float s = 0.f;
        for (int c = 0; c < SEQ / 8 / 8; ++c) {
            bf16x8 t = p[c];
            bf16x8 o;
#pragma unroll
            for (int j = 0; j < 8; ++j) {
                float e = __expf(bf2f(t[j]) - m);
                s += e;
                o[j] = f2bf(e);                       // unnormalized P, in place
            }
            p[c] = o;
        }
        red[row * 8 + seg] = s;
    }
    __syncthreads();
    if (tid < 16) {
        float s = 0.f;
#pragma unroll
        for (int j = 0; j < 8; ++j) s += red[tid * 8 + j];
        rowinv[tid] = 1.0f / s;                      // fold 1/sum into final store
    }
    __syncthreads();

    // ---- Phase 2: H = P @ V, two d-tiles per A-frag load (V is transposed).
    const __bf16* Vbase = Vt + (size_t)b * EMBED * SEQ;
    for (int dt = wave * 2; dt < EMBED / 16; dt += 8) {           // dt, dt+1
        const int nb = dt * 16;
        fx8 acc0 = {0.f,0.f,0.f,0.f,0.f,0.f,0.f,0.f};
        fx8 acc1 = {0.f,0.f,0.f,0.f,0.f,0.f,0.f,0.f};
        const __bf16* vrow0 = Vbase + (size_t)(nb + laneN) * SEQ;
        const __bf16* vrow1 = vrow0 + (size_t)16 * SEQ;
        __builtin_prefetch(vrow0, 0, 1);
#pragma unroll 4
        for (int js = 0; js < SEQ / 32; ++js) {
            bf16x16 A  = loadA(Pl, SEQ, laneN, js * 32, half);
            bf16x16 B0 = loadB(vrow0, js * 32, half);
            bf16x16 B1 = loadB(vrow1, js * 32, half);
            acc0 = __builtin_amdgcn_wmma_f32_16x16x32_bf16(false, A, false, B0,
                                                           (short)0, acc0, false, false);
            acc1 = __builtin_amdgcn_wmma_f32_16x16x32_bf16(false, A, false, B1,
                                                           (short)0, acc1, false, false);
        }
#pragma unroll
        for (int v = 0; v < 8; ++v) {
            const int M = v + 8 * half;
            const float inv = rowinv[M];
            float* orow = out + (size_t)(r0 + M) * EMBED;
            orow[nb + laneN]      = acc0[v] * inv;
            orow[nb + 16 + laneN] = acc1[v] * inv;
        }
    }
}

// ---------------------------------------------------------------------------
extern "C" void kernel_launch(void* const* d_in, const int* in_sizes, int n_in,
                              void* d_out, int out_size, void* d_ws, size_t ws_size,
                              hipStream_t stream) {
    const float* x  = (const float*)d_in[0];
    const float* Wq = (const float*)d_in[1];
    const float* Wk = (const float*)d_in[2];
    const float* Wv = (const float*)d_in[3];

    // Workspace layout (bf16 stored as ushort): Qb | Kb | Vt | Wb.
    unsigned short* Qb = (unsigned short*)d_ws;
    unsigned short* Kb = Qb + (size_t)NROWS * EMBED;
    unsigned short* Vt = Kb + (size_t)NROWS * EMBED;
    unsigned short* Wb = Vt + (size_t)NROWS * EMBED;

    wconv_kernel<<<dim3(3 * WMAT / (256 * 8)), dim3(256), 0, stream>>>(
        Wq, Wk, Wv, Wb);

    qkv_proj_kernel<<<dim3(NROWS / 16), dim3(128), 0, stream>>>(
        x, Wb, Qb, Kb, Vt);

    // ~91 KB dynamic LDS (> default 64 KB cap) — raise the limit.
    (void)hipFuncSetAttribute((const void*)attn_kernel,
                              hipFuncAttributeMaxDynamicSharedMemorySize,
                              ATTN_LDS_BYTES);
    attn_kernel<<<dim3(NROWS / 16), dim3(128), ATTN_LDS_BYTES, stream>>>(
        Qb, Kb, Vt, (float*)d_out);
}
